// POINT_1b_fc_8452495638808
// MI455X (gfx1250) — compile-verified, matrix operand order checked by reading, not run on hardware
//
#include <hip/hip_runtime.h>
#include <hip/hip_bf16.h>

// ---------------------------------------------------------------------------
// Problem constants
// ---------------------------------------------------------------------------
#define NROI 2048
#define FEAT 1024
#define NH   8
#define DK   128
#define SUBD 128

typedef __attribute__((ext_vector_type(8)))  float  v8f;
typedef __attribute__((ext_vector_type(8)))  __bf16 v8bf;
typedef __attribute__((ext_vector_type(16))) __bf16 v16bf;

__device__ __forceinline__ __bf16 to_bf16(float f) { return (__bf16)f; }

// Load one 16x32 bf16 WMMA operand fragment for this lane from a row-major,
// K-contiguous matrix. Layout (ISA 7.12.2, 16-bit A 16x32): lanes 0-15 hold
// K = {kk+0..7, kk+16..23}, lanes 16-31 hold K = {kk+8..15, kk+24..31};
// vector elements e<8 -> first group, e>=8 -> second group. Two b128 loads.
__device__ __forceinline__ v16bf load_frag(const __bf16* base, int ld, int row,
                                           int kk, int half) {
  const __bf16* p = base + (size_t)row * ld + kk + half * 8;
  v8bf lo = *(const v8bf*)(p);
  v8bf hi = *(const v8bf*)(p + 16);
  return __builtin_shufflevector(lo, hi, 0, 1, 2, 3, 4, 5, 6, 7,
                                 8, 9, 10, 11, 12, 13, 14, 15);
}

// Same fragment gather but from an LDS row (K contiguous) -> ds_load_b128 x2.
__device__ __forceinline__ v16bf load_frag_lds(const __bf16* rowp, int kk, int half) {
  const __bf16* p = rowp + kk + half * 8;
  v8bf lo = *(const v8bf*)(p);
  v8bf hi = *(const v8bf*)(p + 16);
  return __builtin_shufflevector(lo, hi, 0, 1, 2, 3, 4, 5, 6, 7,
                                 8, 9, 10, 11, 12, 13, 14, 15);
}

// ---------------------------------------------------------------------------
// K1: fold weights.
//  o in [0,8)  : Mfold[o][d] = Wg[o][d]                bias = bg[o] + src.Wg[o]
//  o in [8,16) : Mfold[o][d] = (Wk[h] @ Wa[h])[d]      bias = bk[h].Wa[h] + ba[h]
//  o in [16,24): Mfold[o][d] = (Wq[h] @ Wa[h])[d]      bias = bq[h].Wa[h]
// ---------------------------------------------------------------------------
__global__ void fold_weights_kernel(const float* __restrict__ src,
                                    const float* __restrict__ Wg,
                                    const float* __restrict__ bg,
                                    const float* __restrict__ Wq,
                                    const float* __restrict__ bq,
                                    const float* __restrict__ Wk,
                                    const float* __restrict__ bk,
                                    const float* __restrict__ Wa,
                                    const float* __restrict__ ba,
                                    float* __restrict__ Mfold,
                                    float* __restrict__ biasF) {
  __shared__ float red[256];
  int o = blockIdx.x;      // 0..23
  int t = threadIdx.x;     // 0..255
  float partial = 0.0f;
  if (o < 8) {
    for (int d = t; d < FEAT; d += 256) {
      float w = Wg[o * FEAT + d];
      Mfold[o * FEAT + d] = w;
      partial += src[d] * w;
    }
  } else {
    int h = (o - 8) & 7;
    const float* W  = (o < 16) ? Wk : Wq;
    const float* bv = (o < 16) ? bk : bq;
    const float* wa = Wa + h * DK;
    for (int d = t; d < FEAT; d += 256) {
      const float* row = W + ((size_t)h * FEAT + d) * DK;
      float u = 0.0f;
      for (int k2 = 0; k2 < DK; ++k2) u = fmaf(row[k2], wa[k2], u);
      Mfold[o * FEAT + d] = u;
    }
    if (t < DK) partial = bv[h * DK + t] * wa[t];
  }
  red[t] = partial;
  __syncthreads();
  for (int st = 128; st > 0; st >>= 1) {
    if (t < st) red[t] += red[t + st];
    __syncthreads();
  }
  if (t == 0) {
    float b = red[0];
    if (o < 8) b += bg[o];
    else if (o < 16) b += ba[o - 8];   // fold +ba into sk
    biasF[o] = b;
  }
}

// ---------------------------------------------------------------------------
// K2: skinny projection  S = face @ Mfold^T  -> g (relu), skba, sq  (f32 VALU)
// ---------------------------------------------------------------------------
__global__ void proj24_kernel(const float* __restrict__ face,
                              const float* __restrict__ Mfold,
                              const float* __restrict__ biasF,
                              float* __restrict__ g,
                              float* __restrict__ skba,
                              float* __restrict__ sqv) {
  int idx = blockIdx.x * blockDim.x + threadIdx.x;   // 0..49151
  int o = idx % 24;
  int n = idx / 24;
  const float* fr = face + (size_t)n * FEAT;
  const float* mr = Mfold + (size_t)o * FEAT;
  float acc = 0.0f;
  for (int d = 0; d < FEAT; ++d) acc = fmaf(fr[d], mr[d], acc);
  acc += biasF[o];
  if (o < 8)       g[o * NROI + n]           = fmaxf(acc, 0.0f);
  else if (o < 16) skba[(o - 8) * NROI + n]  = acc;
  else             sqv[(o - 16) * NROI + n]  = acc;
}

// ---------------------------------------------------------------------------
// K3 family: f32 -> bf16 converts / transposes
// ---------------------------------------------------------------------------
__global__ void cvt_face_kernel(const float* __restrict__ in, __bf16* __restrict__ out) {
  int i = blockIdx.x * blockDim.x + threadIdx.x;
  out[i] = to_bf16(in[i]);
}
// WvT[(h*128+k)][d] = Wv[h][d][k]
__global__ void cvt_wvT_kernel(const float* __restrict__ Wv, __bf16* __restrict__ out) {
  int idx = blockIdx.x * blockDim.x + threadIdx.x;  // 1024*1024
  int row = idx >> 10, d = idx & 1023;
  int h = row >> 7, k = row & 127;
  out[idx] = to_bf16(Wv[((size_t)h * FEAT + d) * DK + k]);
}
// WsT[s][d] = Ws[d][s]
__global__ void cvt_wsT_kernel(const float* __restrict__ Ws, __bf16* __restrict__ out) {
  int idx = blockIdx.x * blockDim.x + threadIdx.x;  // 128*1024
  int s = idx >> 10, d = idx & 1023;
  out[idx] = to_bf16(Ws[(size_t)d * SUBD + s]);
}
// faT[c][r] = fa[r][c]   (c in [0,1024), r in [0,2048))
__global__ void cvt_faT_kernel(const float* __restrict__ fa, __bf16* __restrict__ out) {
  int idx = blockIdx.x * blockDim.x + threadIdx.x;  // 1024*2048
  int c = idx >> 11, r = idx & 2047;
  out[idx] = to_bf16(fa[(size_t)r * FEAT + c]);
}
// featB = bf16(att + face)
__global__ void make_feat_kernel(const float* __restrict__ att,
                                 const float* __restrict__ face,
                                 __bf16* __restrict__ out) {
  int i = blockIdx.x * blockDim.x + threadIdx.x;
  out[i] = to_bf16(att[i] + face[i]);
}

// ---------------------------------------------------------------------------
// K4: generic bf16 WMMA GEMM:  C[M][N] = A[M][K] * Bt[N][K]^T + bias[N]
// One wave per 32x64 tile: 2 A frags x 4 B frags -> 8 WMMA per 6 frag loads.
// ---------------------------------------------------------------------------
__global__ void gemm_bf16_kernel(const __bf16* __restrict__ A,
                                 const __bf16* __restrict__ Bt,
                                 const float* __restrict__ bias,
                                 float* __restrict__ C,
                                 int M, int N, int K) {
  int L = threadIdx.x;           // 0..31
  int r = L & 15, half = L >> 4;
  int mBase = blockIdx.x * 32;
  int nBase = blockIdx.y * 64;
  v8f acc[2][4];
#pragma unroll
  for (int t = 0; t < 2; ++t)
#pragma unroll
    for (int s = 0; s < 4; ++s) acc[t][s] = v8f{};

  for (int kk = 0; kk < K; kk += 32) {
    __builtin_prefetch(A + (size_t)(mBase + r) * K + kk + 64, 0, 1);
    v16bf a0 = load_frag(A, K, mBase + r, kk, half);
    v16bf a1 = load_frag(A, K, mBase + 16 + r, kk, half);
    v16bf b[4];
#pragma unroll
    for (int s = 0; s < 4; ++s) b[s] = load_frag(Bt, K, nBase + s * 16 + r, kk, half);
#pragma unroll
    for (int s = 0; s < 4; ++s) {
      acc[0][s] = __builtin_amdgcn_wmma_f32_16x16x32_bf16(false, a0, false, b[s],
                                                          (short)0, acc[0][s], false, false);
      acc[1][s] = __builtin_amdgcn_wmma_f32_16x16x32_bf16(false, a1, false, b[s],
                                                          (short)0, acc[1][s], false, false);
    }
  }
  // C/D layout: VGPR v, lanes 0-15 -> M=v, lanes 16-31 -> M=v+8; N = lane&15
#pragma unroll
  for (int t = 0; t < 2; ++t) {
#pragma unroll
    for (int s = 0; s < 4; ++s) {
      int col = nBase + s * 16 + r;
      float bb = bias[col];
#pragma unroll
      for (int v = 0; v < 8; ++v) {
        int mrow = mBase + t * 16 + v + half * 8;
        C[(size_t)mrow * N + col] = acc[t][s][v] + bb;
      }
    }
  }
}

// ---------------------------------------------------------------------------
// K5: per-(h,i) row online softmax stats over j: t(i,j) = g_i*relu(skba_i+sq_j)
//     pack[h*2048+i] = {g_i, skba_i, m_i, 1/Z_i}.  One wave per row.
// ---------------------------------------------------------------------------
__global__ void softmax_stats_kernel(const float* __restrict__ g,
                                     const float* __restrict__ skba,
                                     const float* __restrict__ sqv,
                                     float4* __restrict__ pack) {
  int wid  = (blockIdx.x * blockDim.x + threadIdx.x) >> 5;  // 0..16383
  int lane = threadIdx.x & 31;
  int h = wid >> 11;
  int i = wid & 2047;
  float gi = g[h * NROI + i];
  float sb = skba[h * NROI + i];
  const float* sqh = sqv + h * NROI;
  float m = -1e30f, s = 0.0f;
  for (int j = lane; j < NROI; j += 32) {
    float t = gi * fmaxf(sb + sqh[j], 0.0f);
    if (t > m) { s = s * __expf(m - t) + 1.0f; m = t; }
    else       { s += __expf(t - m); }
  }
  for (int off = 16; off > 0; off >>= 1) {
    float m2 = __shfl_xor(m, off, 32);
    float s2 = __shfl_xor(s, off, 32);
    float nm = fmaxf(m, m2);
    s = s * __expf(m - nm) + s2 * __expf(m2 - nm);
    m = nm;
  }
  if (lane == 0) pack[h * NROI + i] = make_float4(gi, sb, m, 1.0f / s);
}

// ---------------------------------------------------------------------------
// K6: attention aggregation  att[j][h*128+k] = sum_i w[h,i,j] * fa[h,i,k]
//  - pack table (32KB) staged into LDS with global_load_async_to_lds_b128
//  - w^T computed ONCE per block into LDS (16x512 bf16 chunks), shared by all
//    8 waves (8x fewer exp than recomputing per wave); A frags via ds_load_b128
//  - B frags from faT (K-contiguous) via global_load_b128
// ---------------------------------------------------------------------------
#define WCHUNK 512

__global__ void attn_out_kernel(const float4* __restrict__ pack,
                                const float* __restrict__ sqv,
                                const __bf16* __restrict__ faT,
                                float* __restrict__ att) {
  __shared__ float4  sp[NROI];          // 32 KB
  __shared__ __bf16  wT[16][WCHUNK];    // 16 KB (w^T chunk, row = local j)
  int h = blockIdx.z;
  int tid = threadIdx.y * 32 + threadIdx.x;

  // --- async stage pack[h] into LDS (never touches VGPR data path) ---
  for (int t = tid; t < NROI; t += 256) {
    unsigned lds_addr = (unsigned)(uintptr_t)(&sp[t]);
    const float4* gp = pack + (size_t)h * NROI + t;
    asm volatile("global_load_async_to_lds_b128 %0, %1, off"
                 :: "v"(lds_addr), "v"(gp) : "memory");
  }
  asm volatile("s_wait_asynccnt 0x0" ::: "memory");
  __syncthreads();

  int L = threadIdx.x, r = L & 15, half = L >> 4;
  int jloc = tid & 15;                       // j handled in the fill phase
  float sqj = sqv[h * NROI + blockIdx.x * 16 + jloc];
  int rowB = h * DK + threadIdx.y * 16 + r;  // faT row = global att column
  v8f acc = {};

  for (int c0 = 0; c0 < NROI; c0 += WCHUNK) {
    __syncthreads();   // previous chunk fully consumed before overwrite
    // fill w^T chunk: (16 j) x (WCHUNK/8 i-blocks), 4 b128 LDS stores/thread
    for (int blk = tid; blk < 16 * (WCHUNK / 8); blk += 256) {
      int ib = blk >> 4;          // i-block within chunk (blk&15 == jloc)
      int i0 = c0 + ib * 8;
      v8bf wv;
#pragma unroll
      for (int u = 0; u < 8; ++u) {
        float4 p = sp[i0 + u];
        float t = p.x * fmaxf(p.y + sqj, 0.0f);
        wv[u] = to_bf16(__expf(t - p.z) * p.w);
      }
      *(v8bf*)&wT[jloc][ib * 8] = wv;
    }
    __syncthreads();
    // consume chunk: A from LDS, B from global
    for (int ii = 0; ii < WCHUNK; ii += 32) {
      v16bf a = load_frag_lds(&wT[r][0], ii, half);
      v16bf b = load_frag(faT, NROI, rowB, c0 + ii, half);
      acc = __builtin_amdgcn_wmma_f32_16x16x32_bf16(false, a, false, b,
                                                    (short)0, acc, false, false);
    }
  }

  int col = h * DK + threadIdx.y * 16 + r;
#pragma unroll
  for (int v = 0; v < 8; ++v) {
    int row = blockIdx.x * 16 + v + half * 8;
    att[(size_t)row * FEAT + col] = acc[v];
  }
}

// ---------------------------------------------------------------------------
// K7: head — eval BN + classifier + softmax prob of class 1. Wave per row.
// ---------------------------------------------------------------------------
__global__ void head_kernel(const float* __restrict__ x,
                            const float* __restrict__ gamma,
                            const float* __restrict__ beta,
                            const float* __restrict__ mean,
                            const float* __restrict__ var,
                            const float* __restrict__ Wc,
                            const float* __restrict__ bc,
                            float* __restrict__ out) {
  int wid  = (blockIdx.x * blockDim.x + threadIdx.x) >> 5;  // row n
  int lane = threadIdx.x & 31;
  if (wid >= NROI) return;
  const float* xr = x + (size_t)wid * SUBD;
  float l0 = 0.0f, l1 = 0.0f;
  for (int k = lane; k < SUBD; k += 32) {
    float sc = gamma[k] * rsqrtf(var[k] + 1e-5f);
    float xb = (xr[k] - mean[k]) * sc + beta[k];
    l0 = fmaf(xb, Wc[k * 2 + 0], l0);
    l1 = fmaf(xb, Wc[k * 2 + 1], l1);
  }
  for (int off = 16; off > 0; off >>= 1) {
    l0 += __shfl_xor(l0, off, 32);
    l1 += __shfl_xor(l1, off, 32);
  }
  if (lane == 0) out[wid] = 1.0f / (1.0f + __expf((l0 + bc[0]) - (l1 + bc[1])));
}

// ---------------------------------------------------------------------------
// Launch
// ---------------------------------------------------------------------------
extern "C" void kernel_launch(void* const* d_in, const int* in_sizes, int n_in,
                              void* d_out, int out_size, void* d_ws, size_t ws_size,
                              hipStream_t stream) {
  const float* face = (const float*)d_in[0];
  const float* src  = (const float*)d_in[1];
  const float* Wg   = (const float*)d_in[2];
  const float* bg   = (const float*)d_in[3];
  const float* Wq   = (const float*)d_in[4];
  const float* bq   = (const float*)d_in[5];
  const float* Wk   = (const float*)d_in[6];
  const float* bk   = (const float*)d_in[7];
  const float* Wa   = (const float*)d_in[8];
  const float* ba   = (const float*)d_in[9];
  const float* Wv   = (const float*)d_in[10];
  const float* bv   = (const float*)d_in[11];   // [8][128] flat == concat column bias
  const float* Ws   = (const float*)d_in[12];
  const float* bs   = (const float*)d_in[13];
  const float* bng  = (const float*)d_in[14];
  const float* bnb  = (const float*)d_in[15];
  const float* bnm  = (const float*)d_in[16];
  const float* bnv  = (const float*)d_in[17];
  const float* Wc   = (const float*)d_in[18];
  const float* bc   = (const float*)d_in[19];
  float* out = (float*)d_out;

  char* ws = (char*)d_ws;
  size_t off = 0;
  auto carve = [&](size_t bytes) -> char* {
    char* p = ws + off;
    off = (off + bytes + 255) & ~(size_t)255;
    return p;
  };
  __bf16* faceB = (__bf16*)carve((size_t)NROI * FEAT * 2);   // 4 MB
  __bf16* WvT   = (__bf16*)carve((size_t)FEAT * FEAT * 2);   // 2 MB
  __bf16* WsT   = (__bf16*)carve((size_t)SUBD * FEAT * 2);   // 256 KB
  float*  Mfold = (float*)carve(24 * FEAT * 4);
  float*  biasF = (float*)carve(24 * 4);
  float*  g     = (float*)carve((size_t)NH * NROI * 4);
  float*  skba  = (float*)carve((size_t)NH * NROI * 4);
  float*  sqv   = (float*)carve((size_t)NH * NROI * 4);
  float4* pack  = (float4*)carve((size_t)NH * NROI * 16);
  float*  fa    = (float*)carve((size_t)NROI * FEAT * 4);    // 8 MB
  __bf16* faT   = (__bf16*)carve((size_t)FEAT * NROI * 2);   // 4 MB
  float*  att   = (float*)carve((size_t)NROI * FEAT * 4);    // 8 MB
  __bf16* featB = (__bf16*)carve((size_t)NROI * FEAT * 2);   // 4 MB
  float*  x     = (float*)carve((size_t)NROI * SUBD * 4);    // 1 MB
  (void)ws_size; (void)n_in; (void)in_sizes; (void)out_size;

  fold_weights_kernel<<<24, 256, 0, stream>>>(src, Wg, bg, Wq, bq, Wk, bk, Wa, ba,
                                              Mfold, biasF);
  proj24_kernel<<<(NROI * 24) / 256, 256, 0, stream>>>(face, Mfold, biasF, g, skba, sqv);
  cvt_face_kernel<<<(NROI * FEAT) / 256, 256, 0, stream>>>(face, faceB);
  cvt_wvT_kernel<<<(FEAT * FEAT) / 256, 256, 0, stream>>>(Wv, WvT);
  cvt_wsT_kernel<<<(SUBD * FEAT) / 256, 256, 0, stream>>>(Ws, WsT);
  // fa = face @ Wv(concat) + bv   : [2048,1024] x [1024,1024]
  gemm_bf16_kernel<<<dim3(NROI / 32, FEAT / 64), 32, 0, stream>>>(
      faceB, WvT, bv, fa, NROI, FEAT, FEAT);
  cvt_faT_kernel<<<(FEAT * NROI) / 256, 256, 0, stream>>>(fa, faT);
  softmax_stats_kernel<<<(NH * NROI * 32) / 256, 256, 0, stream>>>(g, skba, sqv, pack);
  attn_out_kernel<<<dim3(NROI / 16, 1, NH), dim3(32, 8), 0, stream>>>(pack, sqv, faT, att);
  make_feat_kernel<<<(NROI * FEAT) / 256, 256, 0, stream>>>(att, face, featB);
  // x = feat @ Ws + bs : [2048,1024] x [1024,128]
  gemm_bf16_kernel<<<dim3(NROI / 32, SUBD / 64), 32, 0, stream>>>(
      featB, WsT, bs, x, NROI, SUBD, FEAT);
  head_kernel<<<(NROI * 32) / 256, 256, 0, stream>>>(x, bng, bnb, bnm, bnv, Wc, bc, out);
}
